// SimpleRNN_88802743812790
// MI455X (gfx1250) — matrix operand order repaired and checked by reading, not compiled, not run on hardware
//
#include <hip/hip_runtime.h>

#define VOCAB  32000
#define EMBED  256
#define HIDDEN 512
#define NB     64
#define LSEQ   2048

typedef __attribute__((ext_vector_type(16))) __bf16          v16bf;
typedef __attribute__((ext_vector_type(8)))  float           v8f;
typedef __attribute__((ext_vector_type(8)))  unsigned int    v8u;
typedef __attribute__((ext_vector_type(4)))  unsigned int    u32x4;
typedef __attribute__((ext_vector_type(8)))  int             i32x8;
typedef __attribute__((ext_vector_type(4)))  int             i32x4;

__device__ __forceinline__ unsigned short f2bf(float f) {
    __bf16 b = (__bf16)f;                       // native v_cvt (RNE)
    return __builtin_bit_cast(unsigned short, b);
}
__device__ __forceinline__ float bf2f(unsigned short h) {
    return __uint_as_float(((unsigned)h) << 16);
}
__device__ __forceinline__ float fast_tanh(float x) {
    float r;
    asm("v_tanh_f32 %0, %1" : "=v"(r) : "v"(x));  // CDNA5 native trans op
    return r;
}

// ---------------- workspace layout (bytes) ----------------
#define WS_SYNC   0u            // 256 B: grid-sync counter
#define WS_BIAS   256u          // 2048 B: f32 bias_total[512] = b_hh + b_xh
#define WS_H0     4096u         // 64 KB: h buffer 0 (bf16 64x512)
#define WS_H1     69632u        // 64 KB: h buffer 1
#define WS_PWXH   135168u       // 256 KB: W_xh packed bf16 B-fragments
#define WS_PWHH   397312u       // 512 KB: W_hh packed bf16 B-fragments
#define WS_XP     921600u       // 128 MB: xp bf16 [l][n][h]

// Packed B-fragment layout for a KxH (H=512) matrix, tiles of 32(K) x 16(H):
//   tile (kt, ht) base dword = (kt*32 + ht)*256
//   dword index within tile  = lane*8 + i
//   value = pack( W[kt*32 + bk + 2i][ht*16 + col], W[kt*32 + bk + 2i + 1][ht*16 + col] )
//   bk = (lane<16) ? 0 : 16,  col = lane & 15     (contiguous k = bk + s for element s; A uses the same)

__global__ void __launch_bounds__(256)
k_prep(const float* __restrict__ Whh, const float* __restrict__ bhh,
       const float* __restrict__ Wxh, const float* __restrict__ bxh,
       unsigned* __restrict__ pwxh, unsigned* __restrict__ pwhh,
       float* __restrict__ bias) {
    const unsigned NWXH = (EMBED / 32) * (HIDDEN / 16) * 256;   // 65536 dwords
    const unsigned NWHH = (HIDDEN / 32) * (HIDDEN / 16) * 256;  // 131072 dwords
    unsigned t = blockIdx.x * 256u + threadIdx.x;
    if (t < NWXH) {
        unsigned kt = t / (32u * 256u), rem = t % (32u * 256u);
        unsigned ht = rem / 256u, d = rem % 256u;
        unsigned lane = d >> 3, i = d & 7;
        unsigned bk = (lane < 16) ? 0u : 16u, col = lane & 15u;
        unsigned k0 = kt * 32u + bk + 2u * i;
        unsigned lo = f2bf(Wxh[k0 * HIDDEN + ht * 16u + col]);
        unsigned hi = f2bf(Wxh[(k0 + 1u) * HIDDEN + ht * 16u + col]);
        pwxh[t] = lo | (hi << 16);
    } else if (t < NWXH + NWHH) {
        unsigned g = t - NWXH;
        unsigned kt = g / (32u * 256u), rem = g % (32u * 256u);
        unsigned ht = rem / 256u, d = rem % 256u;
        unsigned lane = d >> 3, i = d & 7;
        unsigned bk = (lane < 16) ? 0u : 16u, col = lane & 15u;
        unsigned k0 = kt * 32u + bk + 2u * i;
        unsigned lo = f2bf(Whh[k0 * HIDDEN + ht * 16u + col]);
        unsigned hi = f2bf(Whh[(k0 + 1u) * HIDDEN + ht * 16u + col]);
        pwhh[g] = lo | (hi << 16);
    } else if (t < NWXH + NWHH + HIDDEN) {
        unsigned j = t - NWXH - NWHH;
        bias[j] = bhh[j] + bxh[j];
    }
}

// ---- Kernel A: fused embed + input projection, bf16 WMMA, xp time-major ----
__global__ void __launch_bounds__(256)
k_embed_proj(const int* __restrict__ X, const float* __restrict__ E,
             const unsigned* __restrict__ pwxh, unsigned short* __restrict__ xp) {
    unsigned tid  = threadIdx.x, lane = tid & 31u, w = tid >> 5;
    unsigned tile = blockIdx.x * 8u + w;          // 0 .. 262143
    unsigned mt   = tile >> 5;                    // M tile (16 rows of l*64+n)
    unsigned ht   = tile & 31u;                   // H tile
    unsigned half = lane >> 4, col = lane & 15u;
    unsigned m0   = mt * 16u;
    unsigned row  = m0 + col;                     // this lane's A row
    unsigned l = row >> 6, n = row & 63u;
    unsigned tok = (unsigned)X[n * LSEQ + l];
    const float* er = E + (size_t)tok * EMBED;

    v8f acc = {};
#pragma unroll
    for (int kt = 0; kt < EMBED / 32; ++kt) {
        int kc = kt * 32;
        int r0 = kc + (half ? 16 : 0);
        int r1 = kc + (half ? 24 : 8);
        const float4* p0 = (const float4*)(er + r0);
        const float4* p1 = (const float4*)(er + r1);
        float4 f0 = p0[0], f1 = p0[1], f2 = p1[0], f3 = p1[1];
        v16bf a;
        a[0]  = (__bf16)f0.x; a[1]  = (__bf16)f0.y; a[2]  = (__bf16)f0.z; a[3]  = (__bf16)f0.w;
        a[4]  = (__bf16)f1.x; a[5]  = (__bf16)f1.y; a[6]  = (__bf16)f1.z; a[7]  = (__bf16)f1.w;
        a[8]  = (__bf16)f2.x; a[9]  = (__bf16)f2.y; a[10] = (__bf16)f2.z; a[11] = (__bf16)f2.w;
        a[12] = (__bf16)f3.x; a[13] = (__bf16)f3.y; a[14] = (__bf16)f3.z; a[15] = (__bf16)f3.w;
        const uint4* bp = (const uint4*)(pwxh + (kt * 32u + ht) * 256u + lane * 8u);
        uint4 b0 = bp[0], b1 = bp[1];
        v8u bv;
        bv[0] = b0.x; bv[1] = b0.y; bv[2] = b0.z; bv[3] = b0.w;
        bv[4] = b1.x; bv[5] = b1.y; bv[6] = b1.z; bv[7] = b1.w;
        acc = __builtin_amdgcn_wmma_f32_16x16x32_bf16(
            false, a,
            false, __builtin_bit_cast(v16bf, bv),
            (short)0, acc, false, false);
    }
    unsigned hcol = ht * 16u + col;
#pragma unroll
    for (int j = 0; j < 8; ++j) {
        unsigned r = m0 + (unsigned)j + 8u * half;     // C/D: VGPR j -> M=j (+8 hi half)
        xp[(size_t)r * HIDDEN + hcol] = f2bf(acc[j]);
    }
}

// ---- Kernel B: persistent 8-WG recurrence; W_hh slice in VGPRs; h via TDM->LDS ----
__global__ void __launch_bounds__(256)
k_rnn(const unsigned* __restrict__ pwhh, const float* __restrict__ bias,
      const unsigned short* __restrict__ xp,
      unsigned short* __restrict__ hb0, unsigned short* __restrict__ hb1,
      float* __restrict__ out, unsigned* __restrict__ syncc) {
    __shared__ unsigned short lds_h[NB * HIDDEN];   // 64 KB: current h, bf16

    const unsigned g = blockIdx.x;                  // 8 WGs, 64-col slice each
    const unsigned tid = threadIdx.x, lane = tid & 31u, w = tid >> 5;
    const unsigned half = lane >> 4, col = lane & 15u;
    const unsigned ncol = w & 3u;
    const unsigned h0 = g * 64u + ncol * 16u;       // this wave's output columns
    const unsigned htile = g * 4u + ncol;
    const unsigned m0 = (w >> 2) * 32u;             // two M tiles: m0, m0+16

    // Preload the 16 K-step B fragments of W_hh[:, h0:h0+16] into VGPRs (128/lane)
    v8u bw[16];
#pragma unroll
    for (int kt = 0; kt < 16; ++kt) {
        const uint4* bp = (const uint4*)(pwhh + ((unsigned)kt * 32u + htile) * 256u + lane * 8u);
        uint4 b0 = bp[0], b1 = bp[1];
        v8u bv;
        bv[0] = b0.x; bv[1] = b0.y; bv[2] = b0.z; bv[3] = b0.w;
        bv[4] = b1.x; bv[5] = b1.y; bv[6] = b1.z; bv[7] = b1.w;
        bw[kt] = bv;
    }
    const float biasc = bias[h0 + col];
    const unsigned ldsoff = (unsigned)(size_t)&lds_h[0];   // LDS aperture low 32 bits = offset

    for (int t = 0; t < LSEQ; ++t) {
        const unsigned short* hcur = (t & 1) ? hb1 : hb0;
        unsigned short* hnext      = (t & 1) ? hb0 : hb1;

        // TDM: DMA current h (64 KB) global -> LDS, issued by wave 0 only
        if (w == 0) {
            unsigned long long ga = (unsigned long long)(size_t)hcur;
            u32x4 g0;
            g0[0] = 1u;                                    // count=1 valid descriptor
            g0[1] = ldsoff;                                // lds_addr
            g0[2] = (unsigned)ga;                          // global_addr[31:0]
            g0[3] = ((unsigned)(ga >> 32) & 0x01FFFFFFu) | 0x80000000u; // addr[56:32] | type=2
            i32x8 g1;
            g1[0] = 0x00030000;   // workgroup_mask=0, data_size=3 (8B)
            g1[1] = 0x20000000;   // tensor_dim0[15:0]=8192 in bits[31:16]
            g1[2] = 0x00010000;   // tensor_dim0[31:16]=0, tensor_dim1=1
            g1[3] = 0x20000000;   // tile_dim0=8192 in bits[31:16]
            g1[4] = 0x00000001;   // tile_dim1=1, tile_dim2=0
            g1[5] = 8192;         // tensor_dim0_stride[31:0]
            g1[6] = 0x20000000;   // stride0[47:32]=0, tensor_dim1_stride[15:0]=8192
            g1[7] = 0;            // tensor_dim1_stride[47:16]=0
            i32x4 z4 = {0, 0, 0, 0};
            i32x8 z8 = {0, 0, 0, 0, 0, 0, 0, 0};
            __builtin_amdgcn_tensor_load_to_lds(g0, g1, z4, z4, z8, 0);
            __builtin_amdgcn_s_wait_tensorcnt(0);
        }
        __syncthreads();

        // Two 16x16 output tiles per wave, K=512 (16 bf16 WMMA k-steps each)
        v8f acc0 = {}, acc1 = {};
#pragma unroll
        for (int kt = 0; kt < 16; ++kt) {
            int r0 = kt * 32 + (half ? 16 : 0);
            int r1 = kt * 32 + (half ? 24 : 8);
            unsigned rowa = m0 + col, rowb = m0 + 16u + col;
            uint4 q0 = *(const uint4*)&lds_h[rowa * HIDDEN + r0];
            uint4 q1 = *(const uint4*)&lds_h[rowa * HIDDEN + r1];
            uint4 q2 = *(const uint4*)&lds_h[rowb * HIDDEN + r0];
            uint4 q3 = *(const uint4*)&lds_h[rowb * HIDDEN + r1];
            v8u a0; a0[0]=q0.x; a0[1]=q0.y; a0[2]=q0.z; a0[3]=q0.w;
                    a0[4]=q1.x; a0[5]=q1.y; a0[6]=q1.z; a0[7]=q1.w;
            v8u a1; a1[0]=q2.x; a1[1]=q2.y; a1[2]=q2.z; a1[3]=q2.w;
                    a1[4]=q3.x; a1[5]=q3.y; a1[6]=q3.z; a1[7]=q3.w;
            v16bf bfr = __builtin_bit_cast(v16bf, bw[kt]);
            acc0 = __builtin_amdgcn_wmma_f32_16x16x32_bf16(
                false, __builtin_bit_cast(v16bf, a0), false, bfr, (short)0, acc0, false, false);
            acc1 = __builtin_amdgcn_wmma_f32_16x16x32_bf16(
                false, __builtin_bit_cast(v16bf, a1), false, bfr, (short)0, acc1, false, false);
        }

        const unsigned short* xt = xp + (size_t)t * (NB * HIDDEN);
        const bool last = (t == LSEQ - 1);
#pragma unroll
        for (int j = 0; j < 8; ++j) {
            unsigned ra = m0 + (unsigned)j + 8u * half;
            unsigned rb = ra + 16u;
            unsigned ca = ra * HIDDEN + h0 + col;
            unsigned cb = rb * HIDDEN + h0 + col;
            float va = fast_tanh(acc0[j] + biasc + bf2f(xt[ca]));
            float vb = fast_tanh(acc1[j] + biasc + bf2f(xt[cb]));
            if (last) { out[ca] = va; out[cb] = vb; }
            else      { hnext[ca] = f2bf(va); hnext[cb] = f2bf(vb); }
        }

        // grid-wide step barrier (8 co-resident WGs; monotonic counter, no reset)
        __builtin_amdgcn_fence(__ATOMIC_RELEASE, "agent");
        __syncthreads();
        if (tid == 0) {
            __hip_atomic_fetch_add(syncc, 1u, __ATOMIC_ACQ_REL, __HIP_MEMORY_SCOPE_AGENT);
            unsigned target = (unsigned)(t + 1) * gridDim.x;
            while (__hip_atomic_load(syncc, __ATOMIC_ACQUIRE, __HIP_MEMORY_SCOPE_AGENT) < target)
                __builtin_amdgcn_s_sleep(1);
        }
        __syncthreads();
    }
}

extern "C" void kernel_launch(void* const* d_in, const int* in_sizes, int n_in,
                              void* d_out, int out_size, void* d_ws, size_t ws_size,
                              hipStream_t stream) {
    (void)in_sizes; (void)n_in; (void)out_size; (void)ws_size;
    const int*   X   = (const int*)d_in[0];
    const float* E   = (const float*)d_in[1];
    const float* Whh = (const float*)d_in[2];
    const float* bhh = (const float*)d_in[3];
    const float* Wxh = (const float*)d_in[4];
    const float* bxh = (const float*)d_in[5];
    float* out = (float*)d_out;

    char* ws = (char*)d_ws;
    unsigned*       syncc = (unsigned*)(ws + WS_SYNC);
    float*          bias  = (float*)(ws + WS_BIAS);
    unsigned short* hb0   = (unsigned short*)(ws + WS_H0);
    unsigned short* hb1   = (unsigned short*)(ws + WS_H1);
    unsigned*       pwxh  = (unsigned*)(ws + WS_PWXH);
    unsigned*       pwhh  = (unsigned*)(ws + WS_PWHH);
    unsigned short* xp    = (unsigned short*)(ws + WS_XP);

    // zero sync counter + both h buffers (h0 = zeros)
    (void)hipMemsetAsync(ws, 0, WS_H1 + 65536u, stream);

    const unsigned prep_threads = 65536u + 131072u + 512u;
    k_prep<<<(prep_threads + 255u) / 256u, 256, 0, stream>>>(Whh, bhh, Wxh, bxh, pwxh, pwhh, bias);

    // 131072/16 = 8192 M tiles * 32 H tiles = 262144 wave tiles / 8 waves per block
    k_embed_proj<<<262144 / 8, 256, 0, stream>>>(X, E, pwxh, xp);

    k_rnn<<<8, 256, 0, stream>>>(pwhh, bias, xp, hb0, hb1, out, syncc);
}
